// SIR_80333068305080
// MI455X (gfx1250) — compile-verified, matrix operand order checked by reading, not run on hardware
//
#include <hip/hip_runtime.h>

// ---------------------------------------------------------------------------
// SIR Gumbel-softmax simulation for MI455X (gfx1250, wave32).
// HBM-bound on edge indices: int64 src packed once to u32 (12.8MB/step).
// Gathered infection state is an exact f16
// mirror (values are small integers), double-buffered (old-state race fix).
// Neighbor aggregation = v_wmma_f32_16x16x32_f16 row-sum; both 16-agent
// groups' loads hoisted into one clause for max MLP. Day reduces are folded
// into the following step kernel (ping-pong partials) to halve launch count.
// Threefry key folds (wave-uniform) hoisted to host.
// ---------------------------------------------------------------------------

#define NA      100000                 // N_AGENTS
#define NE      (NA * 32)              // edges
#define DEG     32
#define NSTEP   100
#define BLK     256                    // 8 waves of 32
#define NB      ((NA + BLK - 1) / BLK) // 391
#define OUT_SUS 0
#define OUT_INF 101
#define OUT_REC 202
#define OUT_ST  303

typedef __attribute__((ext_vector_type(16))) _Float16 v16h;
typedef __attribute__((ext_vector_type(8)))  float    v8f;

// ---------------- JAX Threefry-2x32-20 (host + device) ----------------
__host__ __device__ __forceinline__ unsigned rotl32_(unsigned x, int n) {
    return (x << n) | (x >> (32 - n));
}

__host__ __device__ __forceinline__ void threefry2x32(unsigned k0, unsigned k1,
                                                      unsigned& x0, unsigned& x1) {
    const unsigned ks2 = 0x1BD11BDAu ^ k0 ^ k1;
#define TFR(r) { x0 += x1; x1 = rotl32_(x1, r); x1 ^= x0; }
    x0 += k0;  x1 += k1;
    TFR(13) TFR(15) TFR(26) TFR(6)
    x0 += k1;  x1 += ks2 + 1u;
    TFR(17) TFR(29) TFR(16) TFR(24)
    x0 += ks2; x1 += k0 + 2u;
    TFR(13) TFR(15) TFR(26) TFR(6)
    x0 += k0;  x1 += k1 + 3u;
    TFR(17) TFR(29) TFR(16) TFR(24)
    x0 += k1;  x1 += ks2 + 4u;
    TFR(13) TFR(15) TFR(26) TFR(6)
    x0 += ks2; x1 += k0 + 5u;
#undef TFR
}

// fold_in(key(42), d) = threefry((0,42), (0,d))  -- host-side, wave-uniform
static inline void fold_key_host(unsigned d, unsigned& k0, unsigned& k1) {
    unsigned x0 = 0u, x1 = d;
    threefry2x32(0u, 42u, x0, x1);
    k0 = x0; k1 = x1;
}

__device__ __forceinline__ float bits_to_gumbel(unsigned bits) {
    const float TINY = 1.17549435e-38f;
    float u = __uint_as_float((bits >> 9) | 0x3F800000u) - 1.0f;   // [0,1)
    u = u * (1.0f - TINY) + TINY;
    u = fmaxf(u, TINY);                                            // (tiny,1)
    return -logf(-logf(u));
}

// Gumbel pair for agent i, key (k0,k1): JAX counter layout for size 2*NA.
// flat j = 2i+c ; j<NA -> x0-output of ctr (j, j+NA); j>=NA -> x1 of (j-NA, j).
__device__ __forceinline__ void gumbel_pair(unsigned k0, unsigned k1, int i,
                                            float& g0, float& g1) {
    unsigned j0 = 2u * (unsigned)i, j1 = j0 + 1u;
    unsigned b0, b1;
    if (j0 < (unsigned)NA) {               // both j0,j1 < NA (NA even)
        unsigned a0 = j0, a1 = j0 + NA; threefry2x32(k0, k1, a0, a1); b0 = a0;
        unsigned c0 = j1, c1 = j1 + NA; threefry2x32(k0, k1, c0, c1); b1 = c0;
    } else {
        unsigned a0 = j0 - NA, a1 = j0; threefry2x32(k0, k1, a0, a1); b0 = a1;
        unsigned c0 = j1 - NA, c1 = j1; threefry2x32(k0, k1, c0, c1); b1 = c1;
    }
    g0 = bits_to_gumbel(b0);
    g1 = bits_to_gumbel(b1);
}

// Forward value of gumbel_bernoulli == hard Gumbel-max sample.
__device__ __forceinline__ float gb_sample(float p, unsigned k0, unsigned k1, int i) {
    float g0, g1;
    gumbel_pair(k0, k1, i, g0, g1);
    float l0 = logf(p);
    float l1 = logf(1.0f - p);             // p==1 -> -inf -> always sample 1
    return (l0 + g0 >= l1 + g1) ? 1.0f : 0.0f;
}

// Deterministic block tree-reduction of 3 scalars -> partials[bid*3 + c]
__device__ __forceinline__ void block_reduce3(float a, float b, float c,
                                              float* red, float* partials) {
    const int tid = threadIdx.x;
    float v[3] = {a, b, c};
    for (int comp = 0; comp < 3; ++comp) {
        red[tid] = v[comp];
        __syncthreads();
        for (int s = BLK / 2; s > 0; s >>= 1) {
            if (tid < s) red[tid] += red[tid + s];
            __syncthreads();
        }
        if (tid == 0) partials[blockIdx.x * 3 + comp] = red[0];
        __syncthreads();
    }
}

// Deterministic cross-block reduce of previous step's partials -> out[day].
// Runs on warps 0..2 of block 0; barrier-free (stream order guarantees data).
__device__ __forceinline__ void day_reduce(const float* __restrict__ partials_prev,
                                           int day, float* __restrict__ out) {
    const int tid = threadIdx.x;
    const int w = tid >> 5, lane = tid & 31;
    if (blockIdx.x != 0 || w >= 3) return;
    float s = 0.0f;
    for (int b = lane; b < NB; b += 32) s += partials_prev[b * 3 + w];
    for (int off = 16; off > 0; off >>= 1) s += __shfl_down(s, off);
    if (lane == 0) {
        const int o = (w == 0) ? OUT_INF : (w == 1) ? OUT_SUS : OUT_REC;
        out[o + day] = s * (1.0f / (float)NA);
    }
}

__global__ void sir_day_reduce(int day, const float* __restrict__ partials_prev,
                               float* __restrict__ out) {
    day_reduce(partials_prev, day, out);
}

// ---------------- src index pack: int64 -> u32 (halves HBM traffic) --------
__global__ void pack_src(const long long* __restrict__ src64,
                         unsigned* __restrict__ src32) {
    const int i = blockIdx.x * BLK + threadIdx.x;
    if (i < NE) src32[i] = (unsigned)src64[i];
}

// ---------------- t = 0 : x0 from Bernoulli(f0) ----------------
__global__ void sir_init(unsigned k0, unsigned k1, const float* __restrict__ params,
                         float* __restrict__ sus, float* __restrict__ rec,
                         _Float16* __restrict__ infh0,
                         float* __restrict__ partials, float* __restrict__ out) {
    __shared__ float red[BLK];
    const int i = blockIdx.x * BLK + threadIdx.x;
    float iv = 0.0f, sv = 0.0f;
    if (i < NA) {
        float b = gb_sample(params[2], k0, k1, i);   // f0
        iv = b; sv = 1.0f - b;
        infh0[i] = (_Float16)b;                      // exact: b in {0,1}
        sus[i] = sv; rec[i] = 0.0f;
        out[OUT_ST + i] = 1.0f - b;                  // argmax([b,1-b,0])
    }
    block_reduce3(iv, sv, 0.0f, red, partials);
}

// ---------------- one timestep (IdxT = unsigned packed, or long long) ------
template <typename IdxT>
__global__ void sir_step_t(int t, unsigned k0a, unsigned k1a,
                           unsigned k0b, unsigned k1b,
                           const float* __restrict__ params,
                           const IdxT* __restrict__ src,
                           const _Float16* __restrict__ infh_cur,
                           _Float16* __restrict__ infh_next,
                           float* __restrict__ sus, float* __restrict__ rec,
                           const float* __restrict__ partials_prev,
                           float* __restrict__ partials_next,
                           float* __restrict__ out) {
    __shared__ __align__(16) float rowsum[BLK / 32][32];
    __shared__ float red[BLK];

    const int tid  = threadIdx.x;
    const int w    = tid >> 5;
    const int lane = tid & 31;
    const int blockBase = blockIdx.x * BLK;
    const int waveBase  = blockBase + w * 32;
    const int i = blockBase + tid;
    const int ic = (i < NA) ? i : (NA - 1);

    // fold previous step's observable reduce into this launch (block 0 only)
    day_reduce(partials_prev, t, out);

    // hide the later dependent state loads behind the gather phase
    __builtin_prefetch(&sus[ic], 0, 0);              // global_prefetch_b8
    __builtin_prefetch(&rec[ic], 0, 0);

    // ---- neighbor aggregation: nsum[m] = sum_k infh[src[k*NA + agent_m]] ----
    // A (16x32 f16, ISA layout) x ones(32x16) via v_wmma_f32_16x16x32_f16.
    const int  m  = lane & 15;
    const bool hi = (lane >= 16);
    const int  kb = hi ? 8 : 0;                      // K-half offset (A layout)

    v16h ones;
#pragma unroll
    for (int e = 0; e < 16; ++e) ones[e] = (_Float16)1.0f;

    // hoist BOTH groups' index loads (one clause, 32 outstanding) then gathers
    int agentg[2];
#pragma unroll
    for (int g = 0; g < 2; ++g) {
        int a_ = waveBase + g * 16 + m;
        agentg[g] = (a_ < NA) ? a_ : (NA - 1);       // clamp; result never read
    }
    unsigned sidx[2][16];
#pragma unroll
    for (int g = 0; g < 2; ++g)
#pragma unroll
        for (int e = 0; e < 16; ++e) {
            const int K = e + ((e < 8) ? 0 : 8) + kb;
            sidx[g][e] = (unsigned)src[(size_t)K * NA + agentg[g]];
        }
    v16h a[2];
#pragma unroll
    for (int g = 0; g < 2; ++g)
#pragma unroll
        for (int e = 0; e < 16; ++e) a[g][e] = infh_cur[sidx[g][e]];

#pragma unroll
    for (int g = 0; g < 2; ++g) {
        v8f c = {};
        c = __builtin_amdgcn_wmma_f32_16x16x32_f16(
                false, a[g], false, ones, (short)0, c, false, false);
        // D: lane 0 holds rowsums M=0..7, lane 16 holds M=8..15 (col 0)
        if (m == 0) {
            float4* p = (float4*)&rowsum[w][g * 16 + (hi ? 8 : 0)];
            p[0] = make_float4(c[0], c[1], c[2], c[3]);   // ds_store_b128
            p[1] = make_float4(c[4], c[5], c[6], c[7]);
        }
    }
    __syncthreads();
    const float nsum = rowsum[w][lane];

    float iv = 0.0f, sv = 0.0f, rv = 0.0f;
    if (i < NA) {
        const float gamma = params[0], rho = params[1];
        const float in0 = (float)infh_cur[i];        // exact small integer
        const float su0 = sus[i], re0 = rec[i];

        const float n_inf_nb = (1.0f - in0) * nsum;  // msg factored per row
        const float lam  = (su0 + rho * re0) * n_inf_nb * (1.0f / (float)DEG);
        float p_ir = 1.0f - expf(-lam);
        p_ir = fminf(fmaxf(p_ir, 1e-10f), 1.0f);

        const float new_ir  = gb_sample(p_ir, k0a, k1a, i);
        float p_rec = fminf(fmaxf(gamma * in0, 1e-10f), 1.0f);
        const float new_rec = gb_sample(p_rec, k0b, k1b, i);

        const float in2 = in0 + new_ir - new_rec;
        const float su2 = su0 - su0 * new_ir;
        const float re2 = re0 + new_rec - re0 * new_ir;
        infh_next[i] = (_Float16)in2;                // ping-pong: no race
        sus[i] = su2; rec[i] = re2;
        iv = in2; sv = su2; rv = re2;

        int st = 0; float best = in2;                // argmax, first-on-tie
        if (su2 > best) { st = 1; best = su2; }
        if (re2 > best) { st = 2; }
        // streaming, never re-read: non-temporal store
        __builtin_nontemporal_store((float)st,
            &out[OUT_ST + (long long)(t + 1) * NA + i]);
    }
    block_reduce3(iv, sv, rv, red, partials_next);
}

// ---------------------------------------------------------------------------
extern "C" void kernel_launch(void* const* d_in, const int* in_sizes, int n_in,
                              void* d_out, int out_size, void* d_ws, size_t ws_size,
                              hipStream_t stream) {
    const float*     params = (const float*)d_in[0];
    const long long* src64  = (const long long*)d_in[1];   // row 0 of (2, NE)
    float* out = (float*)d_out;
    char*  ws  = (char*)d_ws;

    // ws layout (bytes)
    const size_t OFF_SUS  = 0;
    const size_t OFF_REC  = OFF_SUS + (size_t)NA * 4;
    const size_t OFF_P0   = OFF_REC + (size_t)NA * 4;
    const size_t OFF_P1   = OFF_P0 + (size_t)NB * 3 * 4;
    const size_t OFF_H0   = (OFF_P1 + (size_t)NB * 3 * 4 + 15) & ~(size_t)15;
    const size_t OFF_H1   = (OFF_H0 + (size_t)NA * 2 + 15) & ~(size_t)15;
    const size_t OFF_S32  = (OFF_H1 + (size_t)NA * 2 + 15) & ~(size_t)15;
    const size_t NEED_PACK = OFF_S32 + (size_t)NE * 4;

    float*     sus   = (float*)(ws + OFF_SUS);
    float*     rec   = (float*)(ws + OFF_REC);
    float*     P[2]  = { (float*)(ws + OFF_P0), (float*)(ws + OFF_P1) };
    _Float16*  h0    = (_Float16*)(ws + OFF_H0);
    _Float16*  h1    = (_Float16*)(ws + OFF_H1);
    unsigned*  src32 = (unsigned*)(ws + OFF_S32);

    const bool pack = (ws_size >= NEED_PACK);
    if (pack)
        pack_src<<<(NE + BLK - 1) / BLK, BLK, 0, stream>>>(src64, src32);

    unsigned k0, k1;
    fold_key_host(0u, k0, k1);
    sir_init<<<NB, BLK, 0, stream>>>(k0, k1, params, sus, rec, h0, P[0], out);

    for (int t = 0; t < NSTEP; ++t) {
        unsigned ka0, ka1, kb0, kb1;
        fold_key_host((unsigned)(2 * t + 1), ka0, ka1);
        fold_key_host((unsigned)(2 * t + 2), kb0, kb1);
        _Float16* cur  = (t & 1) ? h1 : h0;
        _Float16* next = (t & 1) ? h0 : h1;
        float* pprev = P[t & 1];          // written by previous kernel
        float* pnext = P[(t + 1) & 1];    // this step's day-(t+1) sums
        if (pack)
            sir_step_t<unsigned><<<NB, BLK, 0, stream>>>(
                t, ka0, ka1, kb0, kb1, params, src32, cur, next,
                sus, rec, pprev, pnext, out);
        else
            sir_step_t<long long><<<NB, BLK, 0, stream>>>(
                t, ka0, ka1, kb0, kb1, params, src64, cur, next,
                sus, rec, pprev, pnext, out);
    }
    // day 100 from the final step's partials
    sir_day_reduce<<<1, 96, 0, stream>>>(NSTEP, P[NSTEP & 1], out);
}
